// SpatialDecoderGMM_45578192945480
// MI455X (gfx1250) — matrix operand order repaired and checked by reading, not compiled, not run on hardware
//
#include <hip/hip_runtime.h>
#include <hip/hip_bf16.h>

#define NN 100000
#define EE 1600000
#define HH 64
#define INCH 161

// ---- workspace layout (float offsets) ----
#define XIN_OFF   0                         // N*64 floats
#define ACC_OFF   (NN * 64)                 // N*128 floats (out_f | out_b)
#define DEGF_OFF  (ACC_OFF + NN * 128)      // N floats
#define DEGB_OFF  (DEGF_OFF + NN)           // N floats
#define PACK_OFF  (DEGB_OFF + NN)           // packed f16 weight fragments
// packed fragment counts (v16h elements): g1: 6*4*32, g2: 4*4*32, g3: 4*6*32
#define PACK2_H (24 * 512)                  // half-element offset of W_filt pack
#define PACK3_H (40 * 512)                  // half-element offset of heads pack

typedef __attribute__((ext_vector_type(16))) _Float16 v16h;
typedef __attribute__((ext_vector_type(8)))  float    v8f;

__device__ __forceinline__ v8f wmma16(v16h a, v16h b, v8f c) {
  return __builtin_amdgcn_wmma_f32_16x16x32_f16(false, a, false, b, (short)0, c,
                                                false, false);
}

__device__ __forceinline__ v16h pack_a8(float4 u0, float4 u1, float4 u2, float4 u3) {
  v16h a;
  a[0] = (_Float16)u0.x;  a[1] = (_Float16)u0.y;  a[2] = (_Float16)u0.z;  a[3] = (_Float16)u0.w;
  a[4] = (_Float16)u1.x;  a[5] = (_Float16)u1.y;  a[6] = (_Float16)u1.z;  a[7] = (_Float16)u1.w;
  a[8] = (_Float16)u2.x;  a[9] = (_Float16)u2.y;  a[10] = (_Float16)u2.z; a[11] = (_Float16)u2.w;
  a[12] = (_Float16)u3.x; a[13] = (_Float16)u3.y; a[14] = (_Float16)u3.z; a[15] = (_Float16)u3.w;
  return a;
}

// ---- zero scratch (acc + deg_f + deg_b contiguous) ----
__global__ void zero_ws_kernel(float4* p, int n4) {
  int i = blockIdx.x * blockDim.x + threadIdx.x;
  int stride = gridDim.x * blockDim.x;
  for (; i < n4; i += stride) p[i] = make_float4(0.f, 0.f, 0.f, 0.f);
}

// ---- pack weights into WMMA B-fragment order (f16) ----
__global__ void pack_weights_kernel(const float* Win, const float* Wfilt,
                                    const float* Wmu, const float* Wsig,
                                    const float* Wpi, _Float16* pack) {
  int tid = blockIdx.x * blockDim.x + threadIdx.x;   // 0..2047
  int lane = tid & 31;
  int frag = tid >> 5;                               // 0..63
  int kb = (lane >> 4) * 8;
  int cl = lane & 15;
  if (frag < 24) {                                   // GEMM1: W_in [161x64] pad K->192
    int s = frag >> 2, t = frag & 3;
    _Float16* dst = pack + (frag * 32 + lane) * 16;
    int col = t * 16 + cl;
    #pragma unroll
    for (int j = 0; j < 16; j++) {
      int k = s * 32 + kb + j + (j & 8);
      dst[j] = (_Float16)((k < INCH) ? Win[k * 64 + col] : 0.f);
    }
  } else if (frag < 40) {                            // GEMM2: W_filt [128x64]
    int f2 = frag - 24;
    int s = f2 >> 2, t = f2 & 3;
    _Float16* dst = pack + PACK2_H + (f2 * 32 + lane) * 16;
    int col = t * 16 + cl;
    #pragma unroll
    for (int j = 0; j < 16; j++) {
      int k = s * 32 + kb + j + (j & 8);
      dst[j] = (_Float16)Wfilt[k * 64 + col];
    }
  } else {                                           // GEMM3: [128x96] = mu|sigma|pi
    int f3 = frag - 40;
    int s = f3 / 6, t = f3 % 6;
    _Float16* dst = pack + PACK3_H + ((s * 6 + t) * 32 + lane) * 16;
    int col = t * 16 + cl;
    const float* W;
    int c;
    if (col < 32)      { W = Wmu;  c = col; }
    else if (col < 64) { W = Wsig; c = col - 32; }
    else               { W = Wpi;  c = col - 64; }
    #pragma unroll
    for (int j = 0; j < 16; j++) {
      int k = s * 32 + kb + j + (j & 8);
      dst[j] = (_Float16)W[k * 32 + c];
    }
  }
}

// ---- concat(x, x_hat_1, h) gather ----
__device__ __forceinline__ float fetch_in(const float* x, const float* xh,
                                          const float* h, int row, int k) {
  if (k == 0) return x[row];
  if (k < 97) return xh[row * 96 + (k - 1)];
  if (k < INCH) return h[row * 64 + (k - 97)];
  return 0.f;
}

// ---- GEMM1: x_in = concat @ W_in + b_in ; one wave -> 16x64 tile ----
__global__ __launch_bounds__(256) void gemm_in_kernel(
    const float* x, const float* xh, const float* h, const float* b_in,
    const _Float16* pack, float* xin) {
  int gw = (blockIdx.x * 256 + threadIdx.x) >> 5;
  if (gw >= NN / 16) return;
  int lane = threadIdx.x & 31;
  int cl = lane & 15, hh = lane >> 4;
  int row = gw * 16 + cl;
  int kb = hh * 8;
  v8f c[4] = {};
  const v16h* pv = (const v16h*)pack;
  for (int s = 0; s < 6; s++) {
    v16h a;
    #pragma unroll
    for (int j = 0; j < 8; j++) {
      a[j]     = (_Float16)fetch_in(x, xh, h, row, s * 32 + kb + j);
      a[j + 8] = (_Float16)fetch_in(x, xh, h, row, s * 32 + 16 + kb + j);
    }
    const v16h* Bp = pv + (s * 4 * 32 + lane);
    #pragma unroll
    for (int t = 0; t < 4; t++) c[t] = wmma16(a, Bp[t * 32], c[t]);
  }
  #pragma unroll
  for (int t = 0; t < 4; t++) {
    float b = b_in[t * 16 + cl];
    #pragma unroll
    for (int r = 0; r < 8; r++) {
      xin[(gw * 16 + r + 8 * hh) * 64 + t * 16 + cl] = c[t][r] + b;
    }
  }
}

// ---- weighted degrees ----
__global__ void deg_kernel(const int* ei, const float* ew, float* degf, float* degb) {
  int e = blockIdx.x * blockDim.x + threadIdx.x;
  if (e >= EE) return;
  int s = ei[e], t = ei[EE + e];
  float w = (s == t) ? 0.f : ew[e];
  if (w != 0.f) {
    atomicAdd(degf + t, w);
    atomicAdd(degb + s, w);
  }
}

// ---- diffusion scatter: one wave per edge, contiguous-lane fp32 atomics (L2) ----
__global__ __launch_bounds__(256) void scatter_kernel(
    const int* ei, const float* ew, const float* degf, const float* degb,
    const float* xin, float* acc) {
  int e = (blockIdx.x * 256 + threadIdx.x) >> 5;
  if (e >= EE) return;
  e = __builtin_amdgcn_readfirstlane(e);   // wave-uniform -> scalar loads below
  int lane = threadIdx.x & 31;
  int s = ei[e], t = ei[EE + e];
  float w = (s == t) ? 0.f : ew[e];
  if (w == 0.f) return;
  float df = degf[t], db = degb[s];
  float wf = w / (df > 0.f ? df : 1.f);
  float wb = w / (db > 0.f ? db : 1.f);
  // columns {lane, lane+32}: every atomic instruction covers one 128B cacheline
  float vs0 = xin[s * 64 + lane];
  float vs1 = xin[s * 64 + lane + 32];
  float vt0 = xin[t * 64 + lane];
  float vt1 = xin[t * 64 + lane + 32];
  atomicAdd(acc + t * 128 + lane,           wf * vs0);
  atomicAdd(acc + t * 128 + lane + 32,      wf * vs1);
  atomicAdd(acc + s * 128 + 64 + lane,      wb * vt0);
  atomicAdd(acc + s * 128 + 64 + lane + 32, wb * vt1);
}

// ---- GEMM2 + GEMM3 + softplus/softmax heads; one wave -> 16-row tile ----
__global__ __launch_bounds__(256) void head_kernel(
    const float* acc, const float* h, const float* b_filt, const float* b_mu,
    const float* b_sig, const float* b_pi, const _Float16* pack,
    float* out_gmm, float* out1) {
  __shared__ __align__(16) float lds[8][16 * 64];
  int gw = (blockIdx.x * 256 + threadIdx.x) >> 5;
  if (gw >= NN / 16) return;
  float* myl = lds[threadIdx.x >> 5];
  int lane = threadIdx.x & 31;
  int cl = lane & 15, hh = lane >> 4;
  int grow = gw * 16;
  int kb = hh * 8;
  const v16h* p2 = (const v16h*)(pack + PACK2_H);
  const v16h* p3 = (const v16h*)(pack + PACK3_H);

  // GEMM2: out = acc[16x128] @ W_filt + b_filt
  v8f o[4] = {};
  for (int s = 0; s < 4; s++) {
    int base = (grow + cl) * 128 + s * 32 + kb;
    float4 u0 = *(const float4*)(acc + base);
    float4 u1 = *(const float4*)(acc + base + 4);
    float4 u2 = *(const float4*)(acc + base + 16);
    float4 u3 = *(const float4*)(acc + base + 20);
    v16h a = pack_a8(u0, u1, u2, u3);
    const v16h* Bp = p2 + (s * 4 * 32 + lane);
    #pragma unroll
    for (int t = 0; t < 4; t++) o[t] = wmma16(a, Bp[t * 32], o[t]);
  }
  #pragma unroll
  for (int t = 0; t < 4; t++) {
    float b = b_filt[t * 16 + cl];
    #pragma unroll
    for (int r = 0; r < 8; r++) {
      float v = o[t][r] + b;
      int m = r + 8 * hh;
      myl[m * 64 + t * 16 + cl] = v;              // stage for GEMM3 A (same wave)
      out1[(grow + m) * 128 + t * 16 + cl] = v;   // out1[:, :64]
    }
  }

  // GEMM3: [out | h] (16x128) @ [W_mu|W_sigma|W_pi] (128x96)
  v8f g[6] = {};
  for (int s = 0; s < 4; s++) {
    float4 u0, u1, u2, u3;
    if (s < 2) {
      int base = cl * 64 + s * 32 + kb;
      u0 = *(const float4*)(myl + base);
      u1 = *(const float4*)(myl + base + 4);
      u2 = *(const float4*)(myl + base + 16);
      u3 = *(const float4*)(myl + base + 20);
    } else {
      int base = (grow + cl) * 64 + (s - 2) * 32 + kb;
      u0 = *(const float4*)(h + base);
      u1 = *(const float4*)(h + base + 4);
      u2 = *(const float4*)(h + base + 16);
      u3 = *(const float4*)(h + base + 20);
    }
    v16h a = pack_a8(u0, u1, u2, u3);
    const v16h* Bp = p3 + (s * 6 * 32 + lane);
    #pragma unroll
    for (int t = 0; t < 6; t++) g[t] = wmma16(a, Bp[t * 32], g[t]);
  }
  // mu bias
  #pragma unroll
  for (int t = 0; t < 2; t++) {
    float b = b_mu[t * 16 + cl];
    #pragma unroll
    for (int r = 0; r < 8; r++) g[t][r] += b;
  }
  // sigma: softplus
  #pragma unroll
  for (int t = 2; t < 4; t++) {
    float b = b_sig[(t - 2) * 16 + cl];
    #pragma unroll
    for (int r = 0; r < 8; r++) {
      float z = g[t][r] + b;
      g[t][r] = (z > 20.f) ? z : log1pf(expf(z));
    }
  }
  // pi: bias then softmax over 32 components (tiles 4,5; 16-lane half groups)
  #pragma unroll
  for (int t = 4; t < 6; t++) {
    float b = b_pi[(t - 4) * 16 + cl];
    #pragma unroll
    for (int r = 0; r < 8; r++) g[t][r] += b;
  }
  #pragma unroll
  for (int r = 0; r < 8; r++) {
    float v4 = g[4][r], v5 = g[5][r];
    float m = fmaxf(v4, v5);
    for (int ofs = 1; ofs < 16; ofs <<= 1) m = fmaxf(m, __shfl_xor(m, ofs, 32));
    float e4 = expf(v4 - m), e5 = expf(v5 - m);
    float sum = e4 + e5;
    for (int ofs = 1; ofs < 16; ofs <<= 1) sum += __shfl_xor(sum, ofs, 32);
    g[4][r] = e4 / sum;
    g[5][r] = e5 / sum;
  }
  // store gmm_out [N,96] = mu|sigma|pi
  #pragma unroll
  for (int t = 0; t < 6; t++)
    #pragma unroll
    for (int r = 0; r < 8; r++)
      out_gmm[(grow + r + 8 * hh) * 96 + t * 16 + cl] = g[t][r];
}

// ---- copy h into out1[:,64:128] and third output ----
__global__ void copy_h_kernel(const float* h, float* out1, float* outh) {
  int i = (blockIdx.x * blockDim.x + threadIdx.x) * 4;
  if (i >= NN * 64) return;
  float4 v = *(const float4*)(h + i);
  *(float4*)(outh + i) = v;
  int row = i >> 6, c = i & 63;
  *(float4*)(out1 + row * 128 + 64 + c) = v;
}

extern "C" void kernel_launch(void* const* d_in, const int* in_sizes, int n_in,
                              void* d_out, int out_size, void* d_ws, size_t ws_size,
                              hipStream_t stream) {
  const float* x      = (const float*)d_in[0];
  const float* xh     = (const float*)d_in[1];
  const float* h      = (const float*)d_in[2];
  const int*   ei     = (const int*)d_in[3];
  const float* ew     = (const float*)d_in[4];
  const float* Win    = (const float*)d_in[5];
  const float* b_in   = (const float*)d_in[6];
  const float* Wfilt  = (const float*)d_in[7];
  const float* b_filt = (const float*)d_in[8];
  const float* Wmu    = (const float*)d_in[9];
  const float* b_mu   = (const float*)d_in[10];
  const float* Wsig   = (const float*)d_in[11];
  const float* b_sig  = (const float*)d_in[12];
  const float* Wpi    = (const float*)d_in[13];
  const float* b_pi   = (const float*)d_in[14];

  float* ws   = (float*)d_ws;
  float* xin  = ws + XIN_OFF;
  float* acc  = ws + ACC_OFF;
  float* degf = ws + DEGF_OFF;
  float* degb = ws + DEGB_OFF;
  _Float16* pack = (_Float16*)(ws + PACK_OFF);

  float* out_gmm = (float*)d_out;
  float* out1    = out_gmm + (size_t)NN * 96;
  float* out_h   = out1 + (size_t)NN * 128;

  // 1. zero acc + degrees (contiguous: N*130 floats)
  zero_ws_kernel<<<2048, 256, 0, stream>>>((float4*)acc, NN * 130 / 4);
  // 2. pack weights into WMMA fragment layout
  pack_weights_kernel<<<8, 256, 0, stream>>>(Win, Wfilt, Wmu, Wsig, Wpi, pack);
  // 3. input projection GEMM (WMMA)
  gemm_in_kernel<<<(NN / 16 + 7) / 8, 256, 0, stream>>>(x, xh, h, b_in, pack, xin);
  // 4. weighted degrees
  deg_kernel<<<(EE + 255) / 256, 256, 0, stream>>>(ei, ew, degf, degb);
  // 5. bidirectional diffusion scatter (L2-resident fp32 atomics)
  scatter_kernel<<<(EE + 7) / 8, 256, 0, stream>>>(ei, ew, degf, degb, xin, acc);
  // 6. filter GEMM + GMM heads (WMMA) + softplus/softmax
  head_kernel<<<(NN / 16 + 7) / 8, 256, 0, stream>>>(acc, h, b_filt, b_mu, b_sig,
                                                     b_pi, pack, out_gmm, out1);
  // 7. h passthrough outputs
  copy_h_kernel<<<(NN * 64 / 4 + 255) / 256, 256, 0, stream>>>(h, out1, out_h);
}
